// NeuralOpinionDynamics_76081050681510
// MI455X (gfx1250) — matrix operand order repaired and checked by reading, not compiled, not run on hardware
//
#include <hip/hip_runtime.h>

typedef __bf16 bf16;
typedef __attribute__((ext_vector_type(16))) bf16  v16bf;
typedef __attribute__((ext_vector_type(8)))  float v8f;

#define N_ROWS 8192
#define DDIM   128

static __device__ __forceinline__ void load8(const float* __restrict__ p, float* f) {
  float4 a = ((const float4*)p)[0];
  float4 b = ((const float4*)p)[1];
  f[0]=a.x; f[1]=a.y; f[2]=a.z; f[3]=a.w;
  f[4]=b.x; f[5]=b.y; f[6]=b.z; f[7]=b.w;
}

static __device__ __forceinline__ v16bf pack16(const float* f) {
  v16bf r;
#pragma unroll
  for (int i = 0; i < 16; ++i) r[i] = (bf16)f[i];
  return r;
}

static __device__ __forceinline__ v8f wmma_bf16(v16bf a, v16bf b, v8f c) {
  return __builtin_amdgcn_wmma_f32_16x16x32_bf16(false, a, false, b, (short)0, c,
                                                 false, false);
}

// B'[k, dcol] = W[dcol, k] (for X @ W^T): two 8-float runs (k=8h.., 16+8h..) of row dcol.
static __device__ __forceinline__ v16bf bfrag_weight(const float* __restrict__ W,
                                                     int dcol, int kt, int h) {
  float f[16];
  const float* p = W + (size_t)dcol * DDIM + kt * 32;
  load8(p + 8 * h, f);
  load8(p + 16 + 8 * h, f + 8);
  return pack16(f);
}

static __device__ __forceinline__ v16bf lds_read_frag(const char* p) {
  union { uint4 q[2]; v16bf v; } u;
  u.q[0] = *(const uint4*)(p);
  u.q[1] = *(const uint4*)(p + 16);
  return u.v;
}

// Async global->LDS copy of one 16KB step-tile (ZWD 8KB | ZWC 8KB), 64B/thread.
// LDS byte offsets assume the dynamic-shared block starts at LDS offset 0
// (only shared allocation in the kernel). Tracked by ASYNCcnt.
static __device__ __forceinline__ void stage_issue(
    const bf16* __restrict__ ZWDp, const bf16* __restrict__ ZWCp,
    int kb, unsigned bufbase, int tid) {
  const bf16* src = (tid < 128) ? ZWDp : ZWCp;
  const int t = tid & 127;
  const unsigned long long g =
      (unsigned long long)(uintptr_t)(src + (size_t)kb * 4096) + (unsigned)t * 64u;
  const unsigned l = bufbase + ((tid < 128) ? 0u : 8192u) + (unsigned)t * 64u;
#pragma unroll
  for (int q = 0; q < 4; ++q) {
    asm volatile("global_load_async_to_lds_b128 %0, %1, off"
                 :: "v"(l + (unsigned)q * 16u), "v"(g + (unsigned)q * 16u)
                 : "memory");
  }
}

static __device__ __forceinline__ void wait_async_all() {
  asm volatile("s_wait_asynccnt 0" ::: "memory");
}

// ---------------------------------------------------------------------------
// Kernel A: s = Z@w_V ; Ej = exp(s) ; ZWDp/ZWCp = bf16(Z@W_{D,C}^T) B-swizzled;
// Hp = bf16(tanh(Z@W1^T + b1)) A-swizzled.
// ---------------------------------------------------------------------------
__global__ __launch_bounds__(256) void nod_prep(
    const float* __restrict__ Z, const float* __restrict__ W_D,
    const float* __restrict__ W_C, const float* __restrict__ wV,
    const float* __restrict__ W1, const float* __restrict__ b1,
    float* __restrict__ s_out, float* __restrict__ Ej,
    bf16* __restrict__ ZWDp, bf16* __restrict__ ZWCp, bf16* __restrict__ Hp)
{
  const int lane = threadIdx.x & 31;
  const int wave = threadIdx.x >> 5;
  const int slab = blockIdx.x * 8 + wave;   // 0..511
  const int r = lane & 15;
  const int h = lane >> 4;
  const int row0 = slab * 16;
  const int row = row0 + r;

  v16bf afr[4];
  float spart = 0.f;
#pragma unroll
  for (int kt = 0; kt < 4; ++kt) {
    float zf[16], wv[16];
    const float* zp = Z + (size_t)row * DDIM + kt * 32;
    load8(zp + 8 * h, zf);
    load8(zp + 16 + 8 * h, zf + 8);
    const float* vp = wV + kt * 32;
    load8(vp + 8 * h, wv);
    load8(vp + 16 + 8 * h, wv + 8);
#pragma unroll
    for (int i = 0; i < 16; ++i) spart += zf[i] * wv[i];
    afr[kt] = pack16(zf);
  }
  spart += __shfl_xor(spart, 16);
  if (lane < 16) {
    s_out[row0 + lane] = spart;
    Ej[row0 + lane] = __expf(spart);
  }

  const v8f vzero = {0.f, 0.f, 0.f, 0.f, 0.f, 0.f, 0.f, 0.f};

#pragma unroll
  for (int m = 0; m < 2; ++m) {
    const float* W = (m == 0) ? W_D : W_C;
    bf16* dst = (m == 0) ? ZWDp : ZWCp;
#pragma unroll
    for (int nt = 0; nt < 8; ++nt) {
      const int dcol = nt * 16 + r;
      v8f acc = vzero;
#pragma unroll
      for (int kt = 0; kt < 4; ++kt)
        acc = wmma_bf16(afr[kt], bfrag_weight(W, dcol, kt, h), acc);
#pragma unroll
      for (int v = 0; v < 8; ++v) {
        const int i = row0 + v + 8 * h;
        const int ktile = i >> 5, rem = i & 31;
        const int hh = (rem >> 3) & 1, idx = (rem & 7) + ((rem >> 4) << 3);
        const int lane2 = hh * 16 + r;
        dst[((size_t)(ktile * 8 + nt) * 32 + lane2) * 16 + idx] = (bf16)acc[v];
      }
    }
  }

#pragma unroll
  for (int nt = 0; nt < 8; ++nt) {
    const int dcol = nt * 16 + r;
    v8f acc = vzero;
#pragma unroll
    for (int kt = 0; kt < 4; ++kt)
      acc = wmma_bf16(afr[kt], bfrag_weight(W1, dcol, kt, h), acc);
    const float bias = b1[dcol];
#pragma unroll
    for (int v = 0; v < 8; ++v) {
      const int i = row0 + v + 8 * h;
      const float hv = tanhf(acc[v] + bias);
      const int it = i >> 4, rr = i & 15;
      const int kt2 = dcol >> 5, rem = dcol & 31;
      const int hh = (rem >> 3) & 1;
      const int idx = (rem & 7) + ((rem >> 4) << 3);
      const int lane2 = hh * 16 + rr;
      Hp[((size_t)(it * 4 + kt2) * 32 + lane2) * 16 + idx] = (bf16)hv;
    }
  }
}

// ---------------------------------------------------------------------------
// Kernel B: rea = H @ W2^T + b2, staged into d_out.
// ---------------------------------------------------------------------------
__global__ __launch_bounds__(256) void nod_rea(
    const bf16* __restrict__ Hp, const float* __restrict__ W2,
    const float* __restrict__ b2, float* __restrict__ out)
{
  const int lane = threadIdx.x & 31;
  const int wave = threadIdx.x >> 5;
  const int slab = blockIdx.x * 8 + wave;
  const int r = lane & 15;
  const int h = lane >> 4;
  const int row0 = slab * 16;

  v16bf afr[4];
#pragma unroll
  for (int kt = 0; kt < 4; ++kt)
    afr[kt] = *(const v16bf*)(Hp + ((size_t)(slab * 4 + kt) * 32 + lane) * 16);

  const v8f vzero = {0.f, 0.f, 0.f, 0.f, 0.f, 0.f, 0.f, 0.f};
#pragma unroll
  for (int nt = 0; nt < 8; ++nt) {
    const int dcol = nt * 16 + r;
    v8f acc = vzero;
#pragma unroll
    for (int kt = 0; kt < 4; ++kt)
      acc = wmma_bf16(afr[kt], bfrag_weight(W2, dcol, kt, h), acc);
    const float bias = b2[dcol];
#pragma unroll
    for (int v = 0; v < 8; ++v)
      out[(size_t)(row0 + v + 8 * h) * DDIM + dcol] = acc[v] + bias;
  }
}

// ---------------------------------------------------------------------------
// Kernel C: fused NxN GEMMs. Wave = 32 rows (two 16-row A tiles share every B
// fragment). Block = 8 waves = 256 rows, one K-slice. B tiles double-buffered
// in LDS via global_load_async_to_lds_b128 (ASYNCcnt). A_norm streamed
// fp32->bf16 in registers; attention weights from precomputed Ej = exp(s_j):
// sigmoid(s_i - s_j) = 1/(1 + Ej*exp(-s_i))  -> fma + v_rcp + v_exp per pair.
// ---------------------------------------------------------------------------
__global__ __launch_bounds__(256) void nod_bigmm(
    const float* __restrict__ A_norm, const float* __restrict__ s,
    const float* __restrict__ Ej,
    const bf16* __restrict__ ZWDp, const bf16* __restrict__ ZWCp,
    float* __restrict__ difP, float* __restrict__ conP, float* __restrict__ denP,
    int ks)
{
  extern __shared__ char smem[];            // 2 x 16KB double buffer
  const int tid = threadIdx.x;
  const int lane = tid & 31;
  const int wave = tid >> 5;
  const int slice = blockIdx.x % ks;
  const int rowblock = blockIdx.x / ks;
  const int r = lane & 15;
  const int h = lane >> 4;
  const int row0 = rowblock * 256 + wave * 32;

  const int kbspan = 256 / ks;
  const int kb0 = slice * kbspan, kb1 = kb0 + kbspan;

  const float ci0 = __expf(-s[row0 + r]);
  const float ci1 = __expf(-s[row0 + 16 + r]);

  v8f accd[2][8], accc[2][8];
  const v8f vzero = {0.f, 0.f, 0.f, 0.f, 0.f, 0.f, 0.f, 0.f};
#pragma unroll
  for (int mt = 0; mt < 2; ++mt)
#pragma unroll
    for (int t = 0; t < 8; ++t) { accd[mt][t] = vzero; accc[mt][t] = vzero; }
  float den0 = 0.f, den1 = 0.f;

  const float* Arow0 = A_norm + (size_t)(row0 + r) * N_ROWS;
  const float* Arow1 = A_norm + (size_t)(row0 + 16 + r) * N_ROWS;

  stage_issue(ZWDp, ZWCp, kb0, 0u, tid);
  wait_async_all();
  __syncthreads();

  for (int kb = kb0; kb < kb1; ++kb) {
    const unsigned cur = (unsigned)((kb - kb0) & 1);
    if (kb + 1 < kb1) stage_issue(ZWDp, ZWCp, kb + 1, (cur ^ 1u) * 16384u, tid);

    const float* ap0 = Arow0 + kb * 32;
    const float* ap1 = Arow1 + kb * 32;
    if (kb + 4 < kb1) {
      __builtin_prefetch(ap0 + 128 + 8 * h, 0, 0);
      __builtin_prefetch(ap1 + 128 + 8 * h, 0, 0);
    }
    float af0[16], af1[16], ej[16], w0[16], w1[16];
    load8(ap0 + 8 * h, af0);
    load8(ap0 + 16 + 8 * h, af0 + 8);
    load8(ap1 + 8 * h, af1);
    load8(ap1 + 16 + 8 * h, af1 + 8);
    const float* ep = Ej + kb * 32;
    load8(ep + 8 * h, ej);
    load8(ep + 16 + 8 * h, ej + 8);
#pragma unroll
    for (int i = 0; i < 16; ++i) {
      const float sg0 = __builtin_amdgcn_rcpf(1.0f + ej[i] * ci0);  // sigmoid
      const float sg1 = __builtin_amdgcn_rcpf(1.0f + ej[i] * ci1);
      const float e0 = __expf(sg0);             // in (1,e): no max-sub needed
      const float e1 = __expf(sg1);
      den0 += e0; den1 += e1;
      w0[i] = e0; w1[i] = e1;
    }
    const v16bf ad0 = pack16(af0);
    const v16bf ad1 = pack16(af1);
    const v16bf ac0 = pack16(w0);
    const v16bf ac1 = pack16(w1);

    const char* buf = smem + cur * 16384u;
#pragma unroll
    for (int nt = 0; nt < 8; ++nt) {
      const v16bf bD = lds_read_frag(buf + nt * 1024 + lane * 32);
      const v16bf bC = lds_read_frag(buf + 8192 + nt * 1024 + lane * 32);
      accd[0][nt] = wmma_bf16(ad0, bD, accd[0][nt]);
      accd[1][nt] = wmma_bf16(ad1, bD, accd[1][nt]);
      accc[0][nt] = wmma_bf16(ac0, bC, accc[0][nt]);
      accc[1][nt] = wmma_bf16(ac1, bC, accc[1][nt]);
    }

    if (kb + 1 < kb1) { wait_async_all(); __syncthreads(); }
  }

  den0 += __shfl_xor(den0, 16);
  den1 += __shfl_xor(den1, 16);
  if (lane < 16) {
    denP[(size_t)slice * N_ROWS + row0 + lane] = den0;
    denP[(size_t)slice * N_ROWS + row0 + 16 + lane] = den1;
  }

  float* dp = difP + (size_t)slice * N_ROWS * DDIM;
  float* cp = conP + (size_t)slice * N_ROWS * DDIM;
#pragma unroll
  for (int mt = 0; mt < 2; ++mt)
#pragma unroll
    for (int v = 0; v < 8; ++v) {
      const int i = row0 + mt * 16 + v + 8 * h;
#pragma unroll
      for (int nt = 0; nt < 8; ++nt) {
        const int d = nt * 16 + r;
        dp[(size_t)i * DDIM + d] = accd[mt][nt][v];
        cp[(size_t)i * DDIM + d] = accc[mt][nt][v];
      }
    }
}

// ---------------------------------------------------------------------------
// Kernel D: combine K-slices, apply ReLU / softmax division / mixing weights.
// ---------------------------------------------------------------------------
__global__ __launch_bounds__(256) void nod_combine(
    const float* __restrict__ difP, const float* __restrict__ conP,
    const float* __restrict__ denP, const float* __restrict__ omega_logit,
    const float* __restrict__ delta_logit, float* __restrict__ out, int ks)
{
  const int idx = blockIdx.x * 256 + threadIdx.x;
  const int i = idx >> 7;
  float dif = 0.f, con = 0.f, den = 0.f;
  for (int sl = 0; sl < ks; ++sl) {
    dif += difP[(size_t)sl * N_ROWS * DDIM + idx];
    con += conP[(size_t)sl * N_ROWS * DDIM + idx];
    den += denP[(size_t)sl * N_ROWS + i];
  }
  const float omega = 1.f / (1.f + __expf(-omega_logit[0]));
  const float delta = 1.f / (1.f + __expf(-delta_logit[0]));
  const float rea = out[idx];
  out[idx] = omega * fmaxf(dif, 0.f) + (1.f - omega) * (con / den) + delta * rea;
}

extern "C" void kernel_launch(void* const* d_in, const int* in_sizes, int n_in,
                              void* d_out, int out_size, void* d_ws, size_t ws_size,
                              hipStream_t stream) {
  const float* Z  = (const float*)d_in[0];
  const float* A  = (const float*)d_in[1];
  const float* WD = (const float*)d_in[2];
  const float* WC = (const float*)d_in[3];
  const float* wV = (const float*)d_in[4];
  const float* W1 = (const float*)d_in[5];
  const float* b1 = (const float*)d_in[6];
  const float* W2 = (const float*)d_in[7];
  const float* b2 = (const float*)d_in[8];
  const float* ol = (const float*)d_in[9];
  const float* dl = (const float*)d_in[10];

  char* ws = (char*)d_ws;
  float* s    = (float*)(ws);                                  // 32 KB
  float* Ej   = (float*)(ws + (32u << 10));                    // 32 KB
  bf16*  ZWDp = (bf16*) (ws + (64u << 10));                    // 2 MB
  bf16*  ZWCp = (bf16*) (ws + (64u << 10) + (2u << 20));       // 2 MB
  bf16*  Hp   = (bf16*) (ws + (64u << 10) + (4u << 20));       // 2 MB

  // K-split chosen deterministically from ws_size (per-slice: 8 MB + 32 KB).
  const size_t base = (64u << 10) + (6u << 20);
  const size_t per_slice = (8u << 20) + (32u << 10);
  int ks = 1;
  if (ws_size >= base + 4 * per_slice) ks = 4;
  else if (ws_size >= base + 2 * per_slice) ks = 2;

  float* difP = (float*)(ws + base);
  float* conP = (float*)(ws + base + (size_t)ks * (4u << 20));
  float* denP = (float*)(ws + base + (size_t)ks * (8u << 20));

  nod_prep<<<64, 256, 0, stream>>>(Z, WD, WC, wV, W1, b1, s, Ej, ZWDp, ZWCp, Hp);
  nod_rea<<<64, 256, 0, stream>>>(Hp, W2, b2, (float*)d_out);
  nod_bigmm<<<32 * ks, 256, 32768, stream>>>(A, s, Ej, ZWDp, ZWCp,
                                             difP, conP, denP, ks);
  nod_combine<<<4096, 256, 0, stream>>>(difP, conP, denP, ol, dl,
                                        (float*)d_out, ks);
}
